// Simple_SelfAttn_82085414961394
// MI455X (gfx1250) — matrix operand order repaired and checked by reading, not compiled, not run on hardware
//
#include <hip/hip_runtime.h>
#include <hip/hip_bf16.h>

// ---------------------------------------------------------------------------
// MI455X (gfx1250) self-attention: f16 WMMA GEMMs + flash-attention softmax.
//   ~550 GFLOP of matmul, ~1.2 GB HBM traffic -> compute bound on
//   v_wmma_f32_16x16x32_f16. QKV GEMM: 128x128 block tile, LDS double
//   buffering via async global->LDS copies (ASYNCcnt). Attention: online
//   softmax in LDS, scores never touch HBM.
// ---------------------------------------------------------------------------

typedef __attribute__((ext_vector_type(16))) _Float16 v16h;
typedef __attribute__((ext_vector_type(8)))  _Float16 v8h;
typedef __attribute__((ext_vector_type(8)))  float    v8f;
typedef __attribute__((ext_vector_type(4)))  float    f32x4;
typedef __attribute__((ext_vector_type(4)))  int      v4i;

// async global->LDS copy operand types: int32x4 in addrspace(1)/(3)
typedef __attribute__((address_space(1))) v4i* gas_v4i_ptr;
typedef __attribute__((address_space(3))) v4i* las_v4i_ptr;

union AFrag { v16h v; v8h h[2]; };

constexpr int kB   = 2;
constexpr int kL   = 2048;
constexpr int kDim = 4096;
constexpr int kH   = 8;
constexpr int kDh  = 512;   // kDim / kH

#if defined(__gfx1250__) && __has_builtin(__builtin_amdgcn_global_load_async_to_lds_b128)
#define HAVE_ASYNC_LDS 1
#else
#define HAVE_ASYNC_LDS 0
#endif

__device__ __forceinline__ v8f wmma_f16(v16h a, v16h b, v8f c) {
  // D = A(16x32) * B(32x16) + C(16x16), f32 accumulate
  return __builtin_amdgcn_wmma_f32_16x16x32_f16(
      /*neg_a=*/false, a, /*neg_b=*/false, b,
      /*c_mod=*/(short)0, c, /*reuse_a=*/false, /*reuse_b=*/false);
}

__device__ __forceinline__ void wait_async_copies() {
#if HAVE_ASYNC_LDS
#if __has_builtin(__builtin_amdgcn_s_wait_asynccnt)
  __builtin_amdgcn_s_wait_asynccnt(0);
#else
  asm volatile("s_wait_asynccnt 0x0" ::: "memory");
#endif
#endif
}

// Copy a 128-row x 32-half panel (8 KB) from global (row stride kDim) to LDS
// (row stride 32). 256 threads x 2 chunks x 16 B.
__device__ __forceinline__ void stage_panel(const _Float16* __restrict__ gsrc,
                                            _Float16* lds, int tid) {
#pragma unroll
  for (int r = 0; r < 2; ++r) {
    const int c = tid + r * 256;
    const int row = c >> 2, ch = (c & 3) * 8;
#if HAVE_ASYNC_LDS
    __builtin_amdgcn_global_load_async_to_lds_b128(
        (gas_v4i_ptr)(gsrc + (size_t)row * kDim + ch),
        (las_v4i_ptr)(lds + row * 32 + ch),
        0, 0);
#else
    *(v8h*)(lds + row * 32 + ch) = *(const v8h*)(gsrc + (size_t)row * kDim + ch);
#endif
  }
}

// ---------------------------------------------------------------------------
// f32 -> f16 conversion, 8 elements / thread, b128 in / b128 out
// ---------------------------------------------------------------------------
__global__ __launch_bounds__(256) void cvt_f32_f16_kernel(
    const float* __restrict__ in, _Float16* __restrict__ out) {
  const int i = (blockIdx.x * 256 + threadIdx.x) * 8;
  f32x4 a = *(const f32x4*)(in + i);
  f32x4 b = *(const f32x4*)(in + i + 4);
  v8h o;
#pragma unroll
  for (int j = 0; j < 4; ++j) {
    o[j]     = (_Float16)a[j];
    o[4 + j] = (_Float16)b[j];
  }
  *(v8h*)(out + i) = o;
}

// ---------------------------------------------------------------------------
// Y = X * W^T + bias    (X: [B*L, DIM] f16, W: [DIM, DIM] f16 row-major)
// Block tile 128x128, 8 waves, each wave 64x32 = 4x2 WMMA tiles, K step 32.
// A/B k-panels double-buffered in LDS via async global->LDS copies.
// W row n IS column n of the B operand -> two 8-half lane loads at k 8*hi and 16+8*hi.
// storeT==0: Y[b,h,l,dh]   (Q,K layout)
// storeT==1: Y[b,h,dh,l]   (V stored pre-transposed for the P*V GEMM)
// ---------------------------------------------------------------------------
__global__ __launch_bounds__(256) void qkv_gemm_kernel(
    const _Float16* __restrict__ X, const _Float16* __restrict__ W,
    const float* __restrict__ bias, _Float16* __restrict__ Y, int storeT) {
  __shared__ _Float16 sA[2][128 * 32];   // 2 x 8 KB
  __shared__ _Float16 sB[2][128 * 32];   // 2 x 8 KB

  const int tid = threadIdx.x;
  const int wv = tid >> 5, ln = tid & 31, lo = ln & 15, hi = ln >> 4;
  const int m0 = blockIdx.x * 128;
  const int n0 = blockIdx.y * 128;
  const int wm = (wv >> 2) * 64;   // wave's 64-row half
  const int wn = (wv & 3) * 32;    // wave's 32-col quarter

  const _Float16* Ag = X + (size_t)m0 * kDim;
  const _Float16* Bg = W + (size_t)n0 * kDim;

  v8f acc[4][2] = {};

  // prologue: stage k=0 panels
  stage_panel(Ag, sA[0], tid);
  stage_panel(Bg, sB[0], tid);
  wait_async_copies();
  __syncthreads();

  int buf = 0;
  for (int k = 0; k < kDim; k += 32) {
    // prefetch next k-panel into the other buffer (overlaps the WMMAs below)
    if (k + 32 < kDim) {
      stage_panel(Ag + k + 32, sA[buf ^ 1], tid);
      stage_panel(Bg + k + 32, sB[buf ^ 1], tid);
    }

    AFrag a[4], b[2];
#pragma unroll
    for (int mt = 0; mt < 4; ++mt) {
      const _Float16* base = &sA[buf][(wm + mt * 16 + lo) * 32];
      a[mt].h[0] = *(const v8h*)(base + hi * 8);        // K 0..7  / 8..15
      a[mt].h[1] = *(const v8h*)(base + hi * 8 + 16);   // K 16..23/ 24..31
    }
#pragma unroll
    for (int nt = 0; nt < 2; ++nt) {
      const _Float16* base = &sB[buf][(wn + nt * 16 + lo) * 32];
      b[nt].h[0] = *(const v8h*)(base + hi * 8);        // K 0..7  / 8..15
      b[nt].h[1] = *(const v8h*)(base + hi * 8 + 16);   // K 16..23/ 24..31
    }
#pragma unroll
    for (int mt = 0; mt < 4; ++mt)
#pragma unroll
      for (int nt = 0; nt < 2; ++nt)
        acc[mt][nt] = wmma_f16(a[mt].v, b[nt].v, acc[mt][nt]);

    wait_async_copies();   // own prefetch done
    __syncthreads();       // everyone's prefetch visible; panel reads finished
    buf ^= 1;
  }

#pragma unroll
  for (int nt = 0; nt < 2; ++nt) {
    const int n = n0 + wn + nt * 16 + lo;
    const float bv = bias[n];
    const int hh = n >> 9, dh = n & (kDh - 1);
#pragma unroll
    for (int mt = 0; mt < 4; ++mt) {
#pragma unroll
      for (int r = 0; r < 8; ++r) {
        const int m = m0 + wm + mt * 16 + 8 * hi + r;   // C layout: M = r + 8*hi
        const int bb = m >> 11, ll = m & (kL - 1);
        const float val = acc[mt][nt][r] + bv;
        const size_t idx = storeT
            ? ((size_t)((bb * kH + hh) * kDh + dh)) * kL + ll
            : ((size_t)((bb * kH + hh) * kL + ll)) * kDh + dh;
        Y[idx] = (_Float16)val;
      }
    }
  }
}

// ---------------------------------------------------------------------------
// Flash attention: block = 32 query rows of one (b,h); iterate 64-key tiles.
//  Phase A: 8 waves each compute one 16x16 S tile (Q from LDS, K from global)
//  Phase B: online softmax (adj bias + mask) -> P f16 in LDS
//  Phase C: each wave owns a 64-wide Dh slice: O = O*scale + P*V  (V^T global)
// ---------------------------------------------------------------------------
__global__ __launch_bounds__(256) void flash_attn_kernel(
    const _Float16* __restrict__ Q,   // [B,H,L,Dh]
    const _Float16* __restrict__ K,   // [B,H,L,Dh]
    const _Float16* __restrict__ VT,  // [B,H,Dh,L]
    const float* __restrict__ adj,    // [B,L,L]
    const int* __restrict__ mask,     // [B,1,L,L]
    float* __restrict__ out) {        // [B,L,DIM]
  __shared__ _Float16 sQ[32 * kDh];   // 32 KB
  __shared__ float    sS[32 * 64];    // 8 KB
  __shared__ _Float16 sP[32 * 64];    // 4 KB
  __shared__ float sM[32], sL[32], sScale[32];

  const int tid = threadIdx.x;
  const int wv = tid >> 5, ln = tid & 31, lo = ln & 15, hi = ln >> 4;
  const int qt = blockIdx.x & 63;
  const int h  = (blockIdx.x >> 6) & 7;
  const int b  = blockIdx.x >> 9;
  const int q0 = qt * 32;

  const _Float16* qbase = Q + (size_t)(b * kH + h) * kL * kDh;
  const _Float16* kbase = K + (size_t)(b * kH + h) * kL * kDh;
  const _Float16* vbase = VT + (size_t)(b * kH + h) * kDh * kL;
  const float*    adjb  = adj + (size_t)b * kL * kL;
  const int*      mskb  = mask + (size_t)b * kL * kL;

  // stage the 32x512 Q tile in LDS once
  for (int i = tid; i < 32 * kDh / 8; i += 256) {
    const int r = (i * 8) / kDh, c = (i * 8) % kDh;
    *(v8h*)(sQ + r * kDh + c) = *(const v8h*)(qbase + (size_t)(q0 + r) * kDh + c);
  }
  if (tid < 32) { sM[tid] = -3.0e38f; sL[tid] = 0.0f; }
  __syncthreads();

  const int mi = wv >> 2, ni = wv & 3;  // S-phase tile coordinates
  const int dh0 = wv * 64;              // PV-phase Dh slice
  const float sc = 0.044194173824159216f;  // 1/sqrt(512)

  v8f o[2][4] = {};

  for (int kv0 = 0; kv0 < kL; kv0 += 64) {
    // ---- Phase A: S tile = Q K^T ----
    {
      v8f s = {};
      const _Float16* krow =
          kbase + (size_t)(kv0 + ni * 16 + lo) * kDh + hi * 8;
      const _Float16* qrow = sQ + (mi * 16 + lo) * kDh;
#pragma unroll 4
      for (int d = 0; d < kDh; d += 32) {
        AFrag a, bf;
        a.h[0]  = *(const v8h*)(qrow + d + hi * 8);
        a.h[1]  = *(const v8h*)(qrow + d + hi * 8 + 16);
        bf.h[0] = *(const v8h*)(krow + d);
        bf.h[1] = *(const v8h*)(krow + d + 16);
        s = wmma_f16(a.v, bf.v, s);
      }
#pragma unroll
      for (int r = 0; r < 8; ++r)
        sS[(mi * 16 + 8 * hi + r) * 64 + ni * 16 + lo] = s[r] * sc;
    }
    __syncthreads();

    // ---- Phase B: online softmax (8 lanes per row, shfl reductions) ----
    {
      const int row = tid >> 3, sub = tid & 7;
      const int qrow = q0 + row;
      float sv[8];
      float mx = -3.0e38f;
#pragma unroll
      for (int j = 0; j < 8; ++j) {
        const int col = sub * 8 + j;
        const int kc = kv0 + col;
        float x = sS[row * 64 + col] + adjb[(size_t)qrow * kL + kc];
        if (mskb[(size_t)qrow * kL + kc] == 0) x = -1.0e6f;
        sv[j] = x;
        mx = fmaxf(mx, x);
      }
#pragma unroll
      for (int off = 4; off > 0; off >>= 1)
        mx = fmaxf(mx, __shfl_xor(mx, off, 8));
      const float m_old = sM[row];
      const float m_new = fmaxf(m_old, mx);
      float sum = 0.0f;
#pragma unroll
      for (int j = 0; j < 8; ++j) {
        const float p = __expf(sv[j] - m_new);
        sum += p;
        sP[row * 64 + sub * 8 + j] = (_Float16)p;
      }
#pragma unroll
      for (int off = 4; off > 0; off >>= 1)
        sum += __shfl_xor(sum, off, 8);
      if (sub == 0) {
        const float scale = __expf(m_old - m_new);
        sM[row] = m_new;
        sL[row] = sL[row] * scale + sum;
        sScale[row] = scale;
      }
    }
    __syncthreads();

    // ---- Phase C: O = O*scale + P*V ----
#pragma unroll
    for (int mt = 0; mt < 2; ++mt)
#pragma unroll
      for (int r = 0; r < 8; ++r) {
        const float f = sScale[mt * 16 + 8 * hi + r];
#pragma unroll
        for (int nj = 0; nj < 4; ++nj) o[mt][nj][r] *= f;
      }
#pragma unroll
    for (int kk = 0; kk < 64; kk += 32) {
      AFrag a[2];
#pragma unroll
      for (int mt = 0; mt < 2; ++mt) {
        a[mt].h[0] = *(const v8h*)(sP + (mt * 16 + lo) * 64 + kk + hi * 8);
        a[mt].h[1] = *(const v8h*)(sP + (mt * 16 + lo) * 64 + kk + hi * 8 + 16);
      }
#pragma unroll
      for (int nj = 0; nj < 4; ++nj) {
        AFrag bf;
        const _Float16* vrow =
            vbase + (size_t)(dh0 + nj * 16 + lo) * kL + kv0 + kk + hi * 8;
        bf.h[0] = *(const v8h*)(vrow);
        bf.h[1] = *(const v8h*)(vrow + 16);
#pragma unroll
        for (int mt = 0; mt < 2; ++mt)
          o[mt][nj] = wmma_f16(a[mt].v, bf.v, o[mt][nj]);
      }
    }
    // next Phase A's barrier orders sP/sScale reuse; no extra barrier needed
  }

  // ---- epilogue: divide by softmax denominator, write context ----
#pragma unroll
  for (int mt = 0; mt < 2; ++mt)
#pragma unroll
    for (int r = 0; r < 8; ++r) {
      const int row = mt * 16 + 8 * hi + r;
      const float inv = 1.0f / sL[row];
      const int lrow = q0 + row;
#pragma unroll
      for (int nj = 0; nj < 4; ++nj) {
        const int col = h * kDh + dh0 + nj * 16 + lo;
        out[(size_t)(b * kL + lrow) * kDim + col] = o[mt][nj][r] * inv;
      }
    }
}

// ---------------------------------------------------------------------------
extern "C" void kernel_launch(void* const* d_in, const int* in_sizes, int n_in,
                              void* d_out, int out_size, void* d_ws, size_t ws_size,
                              hipStream_t stream) {
  const float* input_emb = (const float*)d_in[0];
  const int*   mask      = (const int*)d_in[1];
  const float* adj       = (const float*)d_in[2];
  const float* Wq        = (const float*)d_in[3];
  const float* bq        = (const float*)d_in[4];
  const float* Wk        = (const float*)d_in[5];
  const float* bk        = (const float*)d_in[6];
  const float* Wv        = (const float*)d_in[7];
  const float* bv        = (const float*)d_in[8];
  float* out = (float*)d_out;

  const size_t MB32 = (size_t)32 * 1024 * 1024;
  char* ws = (char*)d_ws;
  _Float16* x_h = (_Float16*)(ws);             // [B*L, DIM] f16
  _Float16* w_h = (_Float16*)(ws + 1 * MB32);  // [DIM, DIM] f16 (reused x3)
  _Float16* q_h = (_Float16*)(ws + 2 * MB32);  // [B,H,L,Dh] f16
  _Float16* k_h = (_Float16*)(ws + 3 * MB32);  // [B,H,L,Dh] f16
  _Float16* v_t = (_Float16*)(ws + 4 * MB32);  // [B,H,Dh,L] f16 (transposed)

  const int NELT = kB * kL * kDim;  // == kDim*kDim == 16777216
  dim3 cvtGrid(NELT / (256 * 8));
  dim3 gemmGrid(kB * kL / 128, kDim / 128);
  dim3 flashGrid(kB * kH * (kL / 32));
  dim3 blk(256);

  hipLaunchKernelGGL(cvt_f32_f16_kernel, cvtGrid, blk, 0, stream, input_emb, x_h);

  hipLaunchKernelGGL(cvt_f32_f16_kernel, cvtGrid, blk, 0, stream, Wq, w_h);
  hipLaunchKernelGGL(qkv_gemm_kernel, gemmGrid, blk, 0, stream, x_h, w_h, bq, q_h, 0);

  hipLaunchKernelGGL(cvt_f32_f16_kernel, cvtGrid, blk, 0, stream, Wk, w_h);
  hipLaunchKernelGGL(qkv_gemm_kernel, gemmGrid, blk, 0, stream, x_h, w_h, bk, k_h, 0);

  hipLaunchKernelGGL(cvt_f32_f16_kernel, cvtGrid, blk, 0, stream, Wv, w_h);
  hipLaunchKernelGGL(qkv_gemm_kernel, gemmGrid, blk, 0, stream, x_h, w_h, bv, v_t, 1);

  hipLaunchKernelGGL(flash_attn_kernel, flashGrid, blk, 0, stream,
                     q_h, k_h, v_t, adj, mask, out);
}
